// SoftMaskedCrossAttention_15418932592805
// MI455X (gfx1250) — matrix operand order repaired and checked by reading, not compiled, not run on hardware
//
#include <hip/hip_runtime.h>
#include <hip/hip_bf16.h>

#define D_MODEL 512
#define N_HEADS 8
#define D_KH    64
#define BATCH   4
#define L_Q     2048
#define L_KV    2048

typedef __attribute__((ext_vector_type(16))) __bf16 v16bf;
typedef __attribute__((ext_vector_type(8)))  float  v8f;
typedef __attribute__((ext_vector_type(4)))  unsigned int v4u;
typedef __attribute__((ext_vector_type(8)))  int v8i;
typedef __attribute__((ext_vector_type(4)))  int v4i;

__device__ __forceinline__ v8f wmma_bf16(v16bf a, v16bf b, v8f c) {
    // 8 args: (neg_a, A, neg_b, B, c_mod, C, reuse_a, reuse_b)
    return __builtin_amdgcn_wmma_f32_16x16x32_bf16(false, a, false, b, (short)0, c, false, false);
}

// ---- Async global->LDS copy of 16 bytes per lane (ASYNCcnt-tracked) ----
__device__ __forceinline__ void async_copy_b128(unsigned lds_off, const void* gaddr) {
    asm volatile("global_load_async_to_lds_b128 %0, %1, off"
                 :: "v"(lds_off), "v"(gaddr) : "memory");
}
template <int N>
__device__ __forceinline__ void wait_async() {
    asm volatile("s_wait_asynccnt %0" :: "n"(N) : "memory");
}

// ---- Tensor Data Mover: DMA one 32(rows) x 64(d) bf16 tile (row stride 512
// elems) from global into contiguous LDS. TENSORcnt-tracked. ----
__device__ __forceinline__ void tdm_load_tile_32x64(unsigned lds_addr, const void* gaddr) {
    const unsigned long long ga = (unsigned long long)(uintptr_t)gaddr;
    v4u g0;
    g0[0] = 1u;                                              // count=1, user mode
    g0[1] = lds_addr;                                        // LDS byte address
    g0[2] = (unsigned)(ga & 0xFFFFFFFFu);                    // global_addr[31:0]
    g0[3] = (unsigned)((ga >> 32) & 0x01FFFFFFu) | (2u << 30); // addr[56:32] | type=2
    v8i g1;
    g1[0] = 0x00010000;            // workgroup_mask=0, data_size=1 (2B), no flags
    g1[1] = (int)(64u << 16);      // tensor_dim0 = 64  (bits 79:48)
    g1[2] = (int)(2048u << 16);    // tensor_dim1 = 2048 (bits 111:80)
    g1[3] = (int)(64u << 16);      // tile_dim0 = 64 (bits 127:112)
    g1[4] = 32;                    // tile_dim1 = 32 (bits 143:128), tile_dim2=0
    g1[5] = 512;                   // tensor_dim0_stride = 512 elems (bits 207:160)
    g1[6] = 0;
    g1[7] = 0;
    const v4i z = {0, 0, 0, 0};
#if defined(__clang_major__) && __clang_major__ >= 23
    const v8i z8 = {0, 0, 0, 0, 0, 0, 0, 0};
    __builtin_amdgcn_tensor_load_to_lds(g0, g1, z, z, z8, 0);
#else
    __builtin_amdgcn_tensor_load_to_lds(g0, g1, z, z, 0);
#endif
}

// ---------------------------------------------------------------------------
// Projection: Y[m,n] = sum_k X[m,k] * W[n,k] + bias[n]   (torch Linear)
// One wave per 64x16 output strip: 4 M-tiles share each W (B) fragment.
// ---------------------------------------------------------------------------
__global__ void proj_f32_bf16(const float* __restrict__ X,
                              const float* __restrict__ W,
                              const float* __restrict__ bias,
                              __bf16* __restrict__ Y) {
    const int lane = threadIdx.x & 31;
    const int hl   = lane >> 4;   // lane half (0/1)
    const int ln   = lane & 15;
    const int m0   = blockIdx.x * 64;
    const int nt   = blockIdx.y;
    const int n    = nt * 16 + ln;

    v8f acc[4] = {{}, {}, {}, {}};
    for (int kk = 0; kk < D_MODEL; kk += 32) {
        v16bf bw;  // B 32x16: lane = N, K = hl*16 + e (contiguous)
        const float* wrow = W + (size_t)n * D_MODEL + kk + hl * 16;
#pragma unroll
        for (int e = 0; e < 16; ++e) bw[e] = (__bf16)wrow[e];
#pragma unroll
        for (int mt = 0; mt < 4; ++mt) {
            v16bf a;  // A 16x32: K = hl*16 + e (contiguous)
            const float* xrow = X + (size_t)(m0 + mt * 16 + ln) * D_MODEL + kk;
#pragma unroll
            for (int e = 0; e < 16; ++e) {
                const int k = hl * 16 + e;
                a[e] = (__bf16)xrow[k];
            }
            acc[mt] = wmma_bf16(a, bw, acc[mt]);
        }
    }
#pragma unroll
    for (int mt = 0; mt < 4; ++mt)
#pragma unroll
        for (int r = 0; r < 8; ++r) {
            const int row = m0 + mt * 16 + r + 8 * hl;   // C/D: VGPR r -> M=r+8*hl
            Y[(size_t)row * D_MODEL + nt * 16 + ln] = (__bf16)(acc[mt][r] + bias[n]);
        }
}

__global__ void proj_bf16_f32(const __bf16* __restrict__ X,
                              const float* __restrict__ W,
                              const float* __restrict__ bias,
                              float* __restrict__ Y) {
    const int lane = threadIdx.x & 31;
    const int hl   = lane >> 4;
    const int ln   = lane & 15;
    const int m0   = blockIdx.x * 64;
    const int nt   = blockIdx.y;
    const int n    = nt * 16 + ln;

    v8f acc[4] = {{}, {}, {}, {}};
    for (int kk = 0; kk < D_MODEL; kk += 32) {
        v16bf bw;
        const float* wrow = W + (size_t)n * D_MODEL + kk + hl * 16;
#pragma unroll
        for (int e = 0; e < 16; ++e) bw[e] = (__bf16)wrow[e];
#pragma unroll
        for (int mt = 0; mt < 4; ++mt) {
            v16bf a;
            const __bf16* xrow = X + (size_t)(m0 + mt * 16 + ln) * D_MODEL + kk;
#pragma unroll
            for (int e = 0; e < 16; ++e) {
                const int k = hl * 16 + e;
                a[e] = xrow[k];
            }
            acc[mt] = wmma_bf16(a, bw, acc[mt]);
        }
    }
#pragma unroll
    for (int mt = 0; mt < 4; ++mt)
#pragma unroll
        for (int r = 0; r < 8; ++r) {
            const int row = m0 + mt * 16 + r + 8 * hl;
            Y[(size_t)row * D_MODEL + nt * 16 + ln] = acc[mt][r] + bias[n];
        }
}

// ---------------------------------------------------------------------------
// Fused attention, per (b,h, 16-row q block). 4 waves; each wave owns 512 of
// the 2048 key columns as 16 chunks of 32 keys. K chunks are DMA'd to LDS by
// the Tensor Data Mover (double-buffered on TENSORcnt); V chunks use
// GLOBAL_LOAD_ASYNC_TO_LDS_B128 (double-buffered on ASYNCcnt). scores
// (16x2048 f32) live in LDS; gated attn streamed to HBM once (non-temporal).
// ---------------------------------------------------------------------------
#define SCORES_FLOATS  (16 * L_KV)                 // 32768 f  (128 KB)
#define OACC_FLOATS    (4 * 16 * 64)               //  4096 f  ( 16 KB)
#define RED_FLOATS     (16 * 8)
#define STAGE_OFF      ((SCORES_FLOATS + OACC_FLOATS + RED_FLOATS + 32) * 4)
#define STAGE_BYTES    (4 * 2 * 4096)              // 4 waves x 2 bufs x 32x64 bf16
#define SMEM_BYTES     (STAGE_OFF + STAGE_BYTES)   // ~177 KB < 320 KB/WGP

__global__ void attn_kernel(const __bf16* __restrict__ Qp,
                            const __bf16* __restrict__ Kp,
                            const __bf16* __restrict__ Vp,
                            const float* __restrict__ Mmask,
                            const float* __restrict__ logit_alpha,
                            float* __restrict__ attn_out,
                            __bf16* __restrict__ O) {
    extern __shared__ char smem_raw[];
    float* scores = (float*)smem_raw;                      // 16*2048
    float* oacc   = scores + SCORES_FLOATS;                // 4*16*64
    float* red    = oacc + OACC_FLOATS;                    // 16*8
    float* rowmax = red + RED_FLOATS;                      // 16
    float* rowinv = rowmax + 16;                           // 16
    __bf16* stage = (__bf16*)(smem_raw + STAGE_OFF);
    const unsigned stage_base = (unsigned)(uintptr_t)(smem_raw + STAGE_OFF);

    const int tid  = threadIdx.x;
    const int wave = tid >> 5;
    const int lane = tid & 31;
    const int hl   = lane >> 4;
    const int ln   = lane & 15;

    const int bh    = blockIdx.y;
    const int b     = bh / N_HEADS;
    const int h     = bh % N_HEADS;
    const int qbase = blockIdx.x * 16;

    // double-buffered per-wave staging: buffer d at bf16 offset wave*4096 + d*2048
    __bf16* wbuf[2] = {stage + wave * 4096, stage + wave * 4096 + 2048};
    const unsigned wbuf_lds[2] = {stage_base + wave * 8192,
                                  stage_base + wave * 8192 + 4096};

    const __bf16* kbase = Kp + (size_t)b * L_KV * D_MODEL + h * D_KH;
    const __bf16* vbase = Vp + (size_t)b * L_KV * D_MODEL + h * D_KH;

    // ---- QK^T: TDM-staged, double buffered ----
    tdm_load_tile_32x64(wbuf_lds[0], kbase + (size_t)(wave * 512) * D_MODEL);

    // load Q A-fragments (16 rows x 64 d, two 32-wide K steps) while TDM runs
    v16bf aq[2];
#pragma unroll
    for (int s = 0; s < 2; ++s) {
        const __bf16* qrow = Qp + ((size_t)b * L_Q + qbase + ln) * D_MODEL + h * D_KH + s * 32;
#pragma unroll
        for (int e = 0; e < 16; ++e) {
            const int d = hl * 16 + e;
            aq[s][e] = qrow[d];
        }
    }

    const float scale = 0.125f;  // 1/sqrt(64)
    for (int c = 0; c < 16; ++c) {
        const int colbase = wave * 512 + c * 32;
        if (c + 1 < 16) {
            tdm_load_tile_32x64(wbuf_lds[(c + 1) & 1],
                                kbase + (size_t)(colbase + 32) * D_MODEL);
            __builtin_amdgcn_s_wait_tensorcnt(1);   // previous DMA complete
        } else {
            __builtin_amdgcn_s_wait_tensorcnt(0);
        }
        const __bf16* kb = wbuf[c & 1];
#pragma unroll
        for (int tt = 0; tt < 2; ++tt) {   // two 16-col tiles per chunk
            v8f acc = {};
#pragma unroll
            for (int s = 0; s < 2; ++s) {
                // B = K^T tile: lane = key col, K = head dim (32B contiguous in LDS)
                const v16bf bk = *(const v16bf*)(kb + (tt * 16 + ln) * 64 + s * 32 + hl * 16);
                acc = wmma_bf16(aq[s], bk, acc);
            }
#pragma unroll
            for (int r = 0; r < 8; ++r)
                scores[(r + 8 * hl) * L_KV + colbase + tt * 16 + ln] = acc[r] * scale;
        }
    }
    __syncthreads();

    // ---- softmax over each of 16 rows (8 threads per row, 256 elems each) ----
    {
        const int row = tid >> 3, seg = tid & 7;
        const float* srow = scores + row * L_KV + seg * 256;
        float mx = -3.402823e38f;
        for (int i = 0; i < 256; ++i) mx = fmaxf(mx, srow[i]);
        red[row * 8 + seg] = mx;
    }
    __syncthreads();
    if (tid < 16) {
        float mx = red[tid * 8];
        for (int i = 1; i < 8; ++i) mx = fmaxf(mx, red[tid * 8 + i]);
        rowmax[tid] = mx;
    }
    __syncthreads();
    {
        const int row = tid >> 3, seg = tid & 7;
        const float mx = rowmax[row];
        const float* srow = scores + row * L_KV + seg * 256;
        float s = 0.0f;
        for (int i = 0; i < 256; ++i) s += __expf(srow[i] - mx);
        red[row * 8 + seg] = s;
    }
    __syncthreads();
    if (tid < 16) {
        float s = 0.0f;
        for (int i = 0; i < 8; ++i) s += red[tid * 8 + i];
        rowinv[tid] = 1.0f / s;
    }
    __syncthreads();

    // ---- gate (post-softmax soft mask) + single streaming write of attn ----
    const float la    = logit_alpha[h];
    const float alpha = 1.0f / (1.0f + __expf(-la));
    const float omal  = 1.0f - alpha;
    float* aout = attn_out + ((size_t)bh * L_Q + qbase) * L_KV;
    const float* mrow = Mmask + ((size_t)h * L_Q + qbase) * L_KV;
    for (int i = tid; i < 16 * L_KV; i += 128) {
        const int row = i >> 11;  // / 2048
        const float g = alpha * __builtin_nontemporal_load(&mrow[i]) + omal;
        const float v = __expf(scores[i] - rowmax[row]) * rowinv[row] * g;
        scores[i] = v;                            // reused as attn@V A-operand
        __builtin_nontemporal_store(v, &aout[i]); // write-once stream, keep L2 for K/V
    }
    __syncthreads();

    // ---- attn @ V: async-B128 staged, double buffered ----
    {
        const char* gv0 = (const char*)(vbase + (size_t)(wave * 512 + lane) * D_MODEL);
#pragma unroll
        for (int j = 0; j < 8; ++j) async_copy_b128(wbuf_lds[0] + lane * 128 + j * 16, gv0 + j * 16);
    }
    v8f oa0 = {}, oa1 = {}, oa2 = {}, oa3 = {};
    for (int c = 0; c < 16; ++c) {
        const int colbase = wave * 512 + c * 32;
        if (c + 1 < 16) {
            const char* gv = (const char*)(vbase + (size_t)(colbase + 32 + lane) * D_MODEL);
            const unsigned dst = wbuf_lds[(c + 1) & 1] + lane * 128;
#pragma unroll
            for (int j = 0; j < 8; ++j) async_copy_b128(dst + j * 16, gv + j * 16);
            wait_async<8>();   // in-order: the previous batch of 8 has landed
        } else {
            wait_async<0>();
        }
        const __bf16* vb = wbuf[c & 1];

        v16bf a;  // A = gated attn tile [16 q x 32 k] from scores
#pragma unroll
        for (int e = 0; e < 16; ++e) {
            const int kc = colbase + hl * 16 + e;
            a[e] = (__bf16)scores[ln * L_KV + kc];
        }
#pragma unroll
        for (int nt = 0; nt < 4; ++nt) {
            v16bf bv;  // B = V tile: lane = head-dim col, K = key row (LDS gather)
#pragma unroll
            for (int e = 0; e < 16; ++e)
                bv[e] = vb[(hl * 16 + e) * 64 + nt * 16 + ln];
            if      (nt == 0) oa0 = wmma_bf16(a, bv, oa0);
            else if (nt == 1) oa1 = wmma_bf16(a, bv, oa1);
            else if (nt == 2) oa2 = wmma_bf16(a, bv, oa2);
            else              oa3 = wmma_bf16(a, bv, oa3);
        }
    }
    // stash per-wave partials
#pragma unroll
    for (int r = 0; r < 8; ++r) {
        const int row = r + 8 * hl;
        oacc[wave * 1024 + row * 64 +  0 + ln] = oa0[r];
        oacc[wave * 1024 + row * 64 + 16 + ln] = oa1[r];
        oacc[wave * 1024 + row * 64 + 32 + ln] = oa2[r];
        oacc[wave * 1024 + row * 64 + 48 + ln] = oa3[r];
    }
    __syncthreads();
    // cross-wave reduce and store O (bf16) for the output projection
    for (int i = tid; i < 16 * 64; i += 128) {
        const float s = oacc[i] + oacc[1024 + i] + oacc[2048 + i] + oacc[3072 + i];
        const int row = i >> 6;
        const int d   = i & 63;
        O[((size_t)b * L_Q + qbase + row) * D_MODEL + h * D_KH + d] = (__bf16)s;
    }
}

// ---------------------------------------------------------------------------
extern "C" void kernel_launch(void* const* d_in, const int* in_sizes, int n_in,
                              void* d_out, int out_size, void* d_ws, size_t ws_size,
                              hipStream_t stream) {
    (void)in_sizes; (void)n_in; (void)out_size; (void)ws_size;

    const float* Q_in  = (const float*)d_in[0];
    const float* K_in  = (const float*)d_in[1];
    const float* V_in  = (const float*)d_in[2];
    const float* Mmask = (const float*)d_in[3];
    const float* Wq    = (const float*)d_in[4];
    const float* bq    = (const float*)d_in[5];
    const float* Wk    = (const float*)d_in[6];
    const float* bk    = (const float*)d_in[7];
    const float* Wv    = (const float*)d_in[8];
    const float* bv    = (const float*)d_in[9];
    const float* Wo    = (const float*)d_in[10];
    const float* bo    = (const float*)d_in[11];
    const float* lal   = (const float*)d_in[12];

    float* out  = (float*)d_out;                                     // [4,2048,512]
    float* attn = out + (size_t)BATCH * L_Q * D_MODEL;               // [4,8,2048,2048]

    const size_t NTOK = (size_t)BATCH * L_Q * D_MODEL;               // 4.19M elems
    __bf16* Qp = (__bf16*)d_ws;
    __bf16* Kp = Qp + NTOK;
    __bf16* Vp = Kp + NTOK;
    __bf16* O  = Vp + NTOK;                                          // total 32 MB bf16

    dim3 gproj(BATCH * L_Q / 64, D_MODEL / 16);   // (128, 32) 64x16 strips
    proj_f32_bf16<<<gproj, 32, 0, stream>>>(Q_in, Wq, bq, Qp);
    proj_f32_bf16<<<gproj, 32, 0, stream>>>(K_in, Wk, bk, Kp);
    proj_f32_bf16<<<gproj, 32, 0, stream>>>(V_in, Wv, bv, Vp);

    dim3 gattn(L_Q / 16, BATCH * N_HEADS);        // (128, 32)
    attn_kernel<<<gattn, 128, SMEM_BYTES, stream>>>(Qp, Kp, Vp, Mmask, lal, attn, O);

    proj_bf16_f32<<<gproj, 32, 0, stream>>>(O, Wo, bo, out);
}
